// DCRNNModel_86079734546628
// MI455X (gfx1250) — compile-verified, matrix operand order checked by reading, not run on hardware
//
#include <hip/hip_runtime.h>
#include <math.h>
#include <stdint.h>

// ---------------------------------------------------------------------------
// DCRNN forward for MI455X (gfx1250, wave32).
// All matmuls (diffusion S@Z and dense Zcat@W) run on v_wmma_f32_16x16x32_bf16
// with fp32 accumulation; sigmoid/tanh/GRU-combine fused into GEMM epilogues.
// K-tiles are software-pipelined through double-buffered LDS; the A tile is
// staged with the CDNA5 async global->LDS engine (ASYNCcnt) when available.
// ---------------------------------------------------------------------------

typedef __bf16 v16bf __attribute__((ext_vector_type(16)));
typedef float  v8f   __attribute__((ext_vector_type(8)));
typedef int    v4i   __attribute__((ext_vector_type(4)));

#define AS1 __attribute__((address_space(1)))
#define AS3 __attribute__((address_space(3)))

#define NP     384   // padded node count (325 -> 384 = 12*32)
#define BSZ    64
#define HDIM   128
#define NNODES 325
#define TSTEPS 12
#define DIN    2

#define BM 128
#define BN 128
#define BK 32
#define LDSP (BK + 8)   // pad LDS rows to 40 elems (80B) to spread banks

#if __has_builtin(__builtin_amdgcn_global_load_async_to_lds_b128)
#define HAVE_ASYNC_LDS 1
#else
#define HAVE_ASYNC_LDS 0
#endif

union ABfrag { uint4 u[2]; v16bf v; };

__device__ __forceinline__ void wait_async_zero()
{
#if HAVE_ASYNC_LDS
#if __has_builtin(__builtin_amdgcn_s_wait_asynccnt)
  __builtin_amdgcn_s_wait_asynccnt(0);
#else
  asm volatile("s_wait_asynccnt 0x0" ::: "memory");
#endif
#endif
}

// Generic bf16 GEMM: D = A(MxK) * B(KxN), fp32 accum.
// A row-major (lda). If slabCp > 0, the K axis is decomposed into 5 slabs of
// slabCp channels, each slab a contiguous [M x slabCp] block at stride slabSS
// (used for the concatenated-diffusion-features matmul).
// mode 0: store bf16 to Dbf            (diffusion)
// mode 1: store sigmoid(acc+bias) f32  (GRU gates -> zr)
// mode 2: hc=tanh(acc+bias); h = z*h + (1-z)*hc  (fused GRU update)
__global__ __launch_bounds__(256) void gemm_bf16_wmma(
    const __bf16* __restrict__ A, long lda, int slabCp, long slabSS,
    const __bf16* __restrict__ B, long ldb, int K, int mode,
    __bf16* __restrict__ Dbf, long lddb,
    float* __restrict__ Df, long lddf,
    const float* __restrict__ bias,
    const float* __restrict__ zbuf, long zld,
    float* __restrict__ hbuf, long hld)
{
  __shared__ __align__(16) __bf16 As[2][BM][LDSP];
  __shared__ __align__(16) __bf16 Bs[2][BN][LDSP];   // transposed: [n][k]

  const int tid  = threadIdx.x;
  const long m0  = (long)blockIdx.y * BM;
  const long n0  = (long)blockIdx.x * BN;
  const int lane = tid & 31, half = lane >> 4, lr = lane & 15;
  const int w = tid >> 5, wm = w & 3, wn = w >> 2;  // 4x2 wave grid
  const int koff = half * 8;

  v8f acc[2][4];
  #pragma unroll
  for (int mt = 0; mt < 2; ++mt)
    #pragma unroll
    for (int nt = 0; nt < 4; ++nt)
      #pragma unroll
      for (int e = 0; e < 8; ++e) acc[mt][nt][e] = 0.0f;

  // Stage one BMxBK A tile and BKxBN B tile (transposed) into LDS buffer `buf`.
  auto stage = [&](int buf, int kt) {
    const __bf16* Ak;
    if (slabCp > 0) {
      int sl = kt / slabCp;                       // BK | slabCp: tile in one slab
      Ak = A + (long)sl * slabSS + (kt - sl * slabCp);
    } else {
      Ak = A + kt;
    }
    const __bf16* Bk = B + (long)kt * ldb + n0;

    // A: 128x32 bf16 in 16B chunks.
    #pragma unroll
    for (int i = 0; i < 2; ++i) {
      int q   = tid + i * 256;
      int row = q >> 2;
      int kc  = (q & 3) << 3;
      const __bf16* gp = Ak + (m0 + row) * lda + kc;
#if HAVE_ASYNC_LDS
      // async global->LDS copy, tracked by ASYNCcnt (no VGPR round-trip)
      __builtin_amdgcn_global_load_async_to_lds_b128(
          (AS1 v4i*)(uintptr_t)gp,
          (AS3 v4i*)(uint32_t)(uintptr_t)&As[buf][row][kc],
          0, 0);
#else
      uint4 v = *reinterpret_cast<const uint4*>(gp);
      *reinterpret_cast<uint4*>(&As[buf][row][kc]) = v;
#endif
    }
    // B: coalesced 16B global reads, scalar transposed LDS writes.
    #pragma unroll
    for (int i = 0; i < 2; ++i) {
      int q  = tid + i * 256;
      int kk = q >> 4;
      int nb = (q & 15) << 3;
      uint4 v = *reinterpret_cast<const uint4*>(Bk + (long)kk * ldb + nb);
      const __bf16* e = reinterpret_cast<const __bf16*>(&v);
      #pragma unroll
      for (int j = 0; j < 8; ++j) Bs[buf][nb + j][kk] = e[j];
    }
    // prefetch the K-tile after this one (global_prefetch_b8)
    if (kt + BK < K) {
      int kn = kt + BK;
      const __bf16* An;
      if (slabCp > 0) { int sl = kn / slabCp; An = A + (long)sl * slabSS + (kn - sl * slabCp); }
      else            { An = A + kn; }
      __builtin_prefetch(An + (m0 + (tid >> 2)) * lda + ((tid & 3) << 3), 0, 1);
      __builtin_prefetch(B + (long)(kn + (tid >> 4)) * ldb + n0 + ((tid & 15) << 3), 0, 1);
    }
  };

  stage(0, 0);
  int buf = 0;
  for (int kt = 0; kt < K; kt += BK) {
    wait_async_zero();       // our async fills of `buf` are done
    __syncthreads();         // everyone's fills of `buf` visible; old reads retired

    // Fragments: lane l holds K {koff..koff+7, 16+koff..}, row/col = lr.
    ABfrag af[2], bfr[4];
    #pragma unroll
    for (int mt = 0; mt < 2; ++mt) {
      int row = wm * 32 + mt * 16 + lr;
      af[mt].u[0] = *reinterpret_cast<const uint4*>(&As[buf][row][koff]);
      af[mt].u[1] = *reinterpret_cast<const uint4*>(&As[buf][row][16 + koff]);
    }
    #pragma unroll
    for (int nt = 0; nt < 4; ++nt) {
      int col = wn * 64 + nt * 16 + lr;
      bfr[nt].u[0] = *reinterpret_cast<const uint4*>(&Bs[buf][col][koff]);
      bfr[nt].u[1] = *reinterpret_cast<const uint4*>(&Bs[buf][col][16 + koff]);
    }

    // overlap: kick off next tile's staging before the WMMA burst
    if (kt + BK < K) stage(buf ^ 1, kt + BK);

    #pragma unroll
    for (int mt = 0; mt < 2; ++mt)
      #pragma unroll
      for (int nt = 0; nt < 4; ++nt)
        acc[mt][nt] = __builtin_amdgcn_wmma_f32_16x16x32_bf16(
            false, af[mt].v, false, bfr[nt].v, (short)0, acc[mt][nt], false, false);

    buf ^= 1;
  }

  // Epilogue. C/D layout: VGPR r -> M = base + r + 8*half, N = base + lr.
  #pragma unroll
  for (int mt = 0; mt < 2; ++mt) {
    #pragma unroll
    for (int nt = 0; nt < 4; ++nt) {
      long gn  = n0 + wn * 64 + nt * 16 + lr;
      long gmb = m0 + wm * 32 + mt * 16 + half * 8;
      #pragma unroll
      for (int r = 0; r < 8; ++r) {
        long gm = gmb + r;
        float v = acc[mt][nt][r];
        if (mode == 0) {
          Dbf[gm * lddb + gn] = (__bf16)v;
        } else if (mode == 1) {
          v += bias[gn];
          Df[gm * lddf + gn] = 1.0f / (1.0f + __expf(-v));
        } else {
          v = tanhf(v + bias[gn]);
          float z  = zbuf[gm * zld + gn];
          float ho = hbuf[gm * hld + gn];
          hbuf[gm * hld + gn] = z * ho + (1.0f - z) * v;
        }
      }
    }
  }
}

// Build slab0 = concat(x, (r?)*h) in bf16, padded to Cp channels.
__global__ void concat_kernel(const float* __restrict__ x, long xsn, long xsb, long xoff,
                              int Dx, int nvx,
                              const float* __restrict__ h, const float* __restrict__ r,
                              __bf16* __restrict__ dst, int Cp)
{
  long i = (long)blockIdx.x * blockDim.x + threadIdx.x;
  long total = (long)NP * BSZ * Cp;
  if (i >= total) return;
  int n   = (int)(i / ((long)BSZ * Cp));
  int rem = (int)(i - (long)n * BSZ * Cp);
  int b = rem / Cp, c = rem % Cp;
  float v;
  if (c < Dx) {
    v = (n < nvx) ? x[xoff + (long)n * xsn + (long)b * xsb + c] : 0.0f;
  } else {
    int hc = c - Dx;
    if (hc < HDIM) {
      long ri = (long)n * BSZ + b;
      v = h[ri * HDIM + hc];
      if (r) v *= r[ri * 256 + hc];
    } else {
      v = 0.0f;
    }
  }
  dst[i] = (__bf16)v;
}

// fp32 weights -> bf16, channel-padded: dst row = slab*Cp + c <- src slab*C + c.
__global__ void prep_weight(const float* __restrict__ src, __bf16* __restrict__ dst,
                            int C, int Cp, int ncols)
{
  long i = (long)blockIdx.x * blockDim.x + threadIdx.x;
  long total = 5L * Cp * ncols;
  if (i >= total) return;
  int row = (int)(i / ncols), col = (int)(i % ncols);
  int slab = row / Cp, c = row % Cp;
  float v = (c < C) ? src[(long)(slab * C + c) * ncols + col] : 0.0f;
  dst[i] = (__bf16)v;
}

// supports (2,325,325) fp32 -> (2,384,384) bf16, zero-padded.
__global__ void prep_supports(const float* __restrict__ src, __bf16* __restrict__ dst)
{
  long i = (long)blockIdx.x * blockDim.x + threadIdx.x;
  long total = 2L * NP * NP;
  if (i >= total) return;
  int s = (int)(i / ((long)NP * NP));
  int r = (int)((i / NP) % NP);
  int c = (int)(i % NP);
  float v = (r < NNODES && c < NNODES) ? src[((long)s * NNODES + r) * NNODES + c] : 0.0f;
  dst[i] = (__bf16)v;
}

__global__ void zero_f32(float* __restrict__ p, long n)
{
  long i = (long)blockIdx.x * blockDim.x + threadIdx.x;
  if (i < n) p[i] = 0.0f;
}

// out[b,t,n,:] = h1[n,b,:] @ Wp + bp; also feeds decoder input buffer.
__global__ void proj_kernel(const float* __restrict__ h1, const float* __restrict__ Wp,
                            const float* __restrict__ bp, float* __restrict__ out,
                            float* __restrict__ dec_in, int t)
{
  int i = blockIdx.x * blockDim.x + threadIdx.x;
  if (i >= NP * BSZ) return;
  int n = i >> 6, b = i & 63;
  const float* hp = h1 + (long)i * HDIM;
  float a0 = bp[0], a1 = bp[1];
  #pragma unroll 4
  for (int c = 0; c < HDIM; ++c) {
    float hv = hp[c];
    a0 += hv * Wp[2 * c];
    a1 += hv * Wp[2 * c + 1];
  }
  dec_in[2L * i]     = a0;
  dec_in[2L * i + 1] = a1;
  if (n < NNODES) {
    long o = (((long)b * TSTEPS + t) * NNODES + n) * 2;
    out[o]     = a0;
    out[o + 1] = a1;
  }
}

extern "C" void kernel_launch(void* const* d_in, const int* in_sizes, int n_in,
                              void* d_out, int out_size, void* d_ws, size_t ws_size,
                              hipStream_t stream)
{
  (void)in_sizes; (void)n_in; (void)out_size; (void)ws_size;
  const float* source   = (const float*)d_in[0];
  const float* supports = (const float*)d_in[2];
  // params ordered [enc0, enc1, dec0, dec1]
  const float* Wg_raw[4] = { (const float*)d_in[3],  (const float*)d_in[7],
                             (const float*)d_in[11], (const float*)d_in[15] };
  const float* bg_raw[4] = { (const float*)d_in[4],  (const float*)d_in[8],
                             (const float*)d_in[12], (const float*)d_in[16] };
  const float* Wc_raw[4] = { (const float*)d_in[5],  (const float*)d_in[9],
                             (const float*)d_in[13], (const float*)d_in[17] };
  const float* bc_raw[4] = { (const float*)d_in[6],  (const float*)d_in[10],
                             (const float*)d_in[14], (const float*)d_in[18] };
  const float* Wp = (const float*)d_in[19];
  const float* bp = (const float*)d_in[20];
  float* out = (float*)d_out;

  // ---- carve workspace (~117 MB) ----
  char* p = (char*)d_ws;
  auto carve = [&](size_t bytes) {
    char* r = p; p += (bytes + 255) & ~(size_t)255; return (void*)r;
  };
  static const int CpL[4] = {160, 256, 160, 256};   // padded per-slab channels
  static const int CL[4]  = {130, 256, 130, 256};   // real per-slab channels
  __bf16* supb = (__bf16*)carve(2ULL * NP * NP * 2);
  __bf16* Wgb[4]; __bf16* Wcb[4];
  for (int i = 0; i < 4; ++i) {
    Wgb[i] = (__bf16*)carve((size_t)5 * CpL[i] * 256 * 2);
    Wcb[i] = (__bf16*)carve((size_t)5 * CpL[i] * 128 * 2);
  }
  float*  h0     = (float*)carve((size_t)NP * BSZ * HDIM * 4);
  float*  h1     = (float*)carve((size_t)NP * BSZ * HDIM * 4);
  float*  zr     = (float*)carve((size_t)NP * BSZ * 256 * 4);
  float*  dec_in = (float*)carve((size_t)NP * BSZ * 2 * 4);
  __bf16* feats  = (__bf16*)carve((size_t)5 * NP * BSZ * 256 * 2);

  // ---- prep: bf16 supports + padded bf16 weights, zero states ----
  {
    long tot = 2L * NP * NP;
    prep_supports<<<dim3((tot + 255) / 256), dim3(256), 0, stream>>>(supports, supb);
  }
  for (int i = 0; i < 4; ++i) {
    long tg = 5L * CpL[i] * 256;
    prep_weight<<<dim3((tg + 255) / 256), dim3(256), 0, stream>>>(Wg_raw[i], Wgb[i], CL[i], CpL[i], 256);
    long tc = 5L * CpL[i] * 128;
    prep_weight<<<dim3((tc + 255) / 256), dim3(256), 0, stream>>>(Wc_raw[i], Wcb[i], CL[i], CpL[i], 128);
  }
  {
    long nh = (long)NP * BSZ * HDIM;
    zero_f32<<<dim3((nh + 255) / 256), dim3(256), 0, stream>>>(h0, nh);
    zero_f32<<<dim3((nh + 255) / 256), dim3(256), 0, stream>>>(h1, nh);
    long nd = (long)NP * BSZ * 2;
    zero_f32<<<dim3((nd + 255) / 256), dim3(256), 0, stream>>>(dec_in, nd);
  }

  auto gemm = [&](const __bf16* A, long lda, int slabCp, long slabSS,
                  const __bf16* Bm, long ldb, int M, int N, int K, int mode,
                  __bf16* Dbf, long lddb, float* Df, long lddf,
                  const float* bias, const float* zb, long zld, float* hb, long hld) {
    dim3 g(N / BN, M / BM);
    gemm_bf16_wmma<<<g, dim3(256), 0, stream>>>(A, lda, slabCp, slabSS, Bm, ldb, K, mode,
                                                Dbf, lddb, Df, lddf, bias, zb, zld, hb, hld);
  };

  // One DCGRU cell: h <- cell(x, h) with layer li's weights.
  auto run_cell = [&](int li, const float* x, long xsn, long xsb, long xoff,
                      int Dx, int nvx, float* h) {
    const int  Cp = CpL[li];
    const long NC = (long)BSZ * Cp;       // diffusion GEMM column count
    const long SS = (long)NP * NC;        // slab stride (elements)
    __bf16* s[5];
    for (int k = 0; k < 5; ++k) s[k] = feats + (long)k * SS;
    const __bf16* S0 = supb;
    const __bf16* S1 = supb + (long)NP * NP;
    const long tot = (long)NP * NC;
    const dim3 cgrid((unsigned)((tot + 255) / 256));

    // ---- gate gconv: zr = sigmoid(gconv(cat[x,h]) Wg + bg) ----
    concat_kernel<<<cgrid, dim3(256), 0, stream>>>(x, xsn, xsb, xoff, Dx, nvx, h, nullptr, s[0], Cp);
    gemm(S0, NP, 0, 0, s[0], NC, NP, (int)NC, NP, 0, s[1], NC, nullptr, 0, nullptr, nullptr, 0, nullptr, 0);
    gemm(S0, NP, 0, 0, s[1], NC, NP, (int)NC, NP, 0, s[2], NC, nullptr, 0, nullptr, nullptr, 0, nullptr, 0);
    gemm(S1, NP, 0, 0, s[0], NC, NP, (int)NC, NP, 0, s[3], NC, nullptr, 0, nullptr, nullptr, 0, nullptr, 0);
    gemm(S1, NP, 0, 0, s[3], NC, NP, (int)NC, NP, 0, s[4], NC, nullptr, 0, nullptr, nullptr, 0, nullptr, 0);
    gemm(feats, Cp, Cp, SS, Wgb[li], 256, NP * BSZ, 256, 5 * Cp, 1,
         nullptr, 0, zr, 256, bg_raw[li], nullptr, 0, nullptr, 0);

    // ---- candidate gconv + fused GRU update: h = z*h + (1-z)*tanh(...) ----
    concat_kernel<<<cgrid, dim3(256), 0, stream>>>(x, xsn, xsb, xoff, Dx, nvx, h, zr + HDIM, s[0], Cp);
    gemm(S0, NP, 0, 0, s[0], NC, NP, (int)NC, NP, 0, s[1], NC, nullptr, 0, nullptr, nullptr, 0, nullptr, 0);
    gemm(S0, NP, 0, 0, s[1], NC, NP, (int)NC, NP, 0, s[2], NC, nullptr, 0, nullptr, nullptr, 0, nullptr, 0);
    gemm(S1, NP, 0, 0, s[0], NC, NP, (int)NC, NP, 0, s[3], NC, nullptr, 0, nullptr, nullptr, 0, nullptr, 0);
    gemm(S1, NP, 0, 0, s[3], NC, NP, (int)NC, NP, 0, s[4], NC, nullptr, 0, nullptr, nullptr, 0, nullptr, 0);
    gemm(feats, Cp, Cp, SS, Wcb[li], 128, NP * BSZ, 128, 5 * Cp, 2,
         nullptr, 0, nullptr, 0, bc_raw[li], zr, 256, h, 128);
  };

  // ---- encoder (stacked layers interleaved over time; equivalent to scan) ----
  for (int t = 0; t < TSTEPS; ++t) {
    run_cell(0, source, 2, (long)TSTEPS * NNODES * 2, (long)t * NNODES * 2, DIN, NNODES, h0);
    run_cell(1, h0, (long)BSZ * HDIM, HDIM, 0, HDIM, NP, h1);
  }
  // ---- decoder (states carry over from encoder finals; go = zeros) ----
  for (int t = 0; t < TSTEPS; ++t) {
    run_cell(2, dec_in, (long)BSZ * 2, 2, 0, DIN, NP, h0);
    run_cell(3, h0, (long)BSZ * HDIM, HDIM, 0, HDIM, NP, h1);
    proj_kernel<<<dim3((NP * BSZ + 255) / 256), dim3(256), 0, stream>>>(h1, Wp, bp, out, dec_in, t);
  }
}
